// Attention_75694503624805
// MI455X (gfx1250) — matrix operand order repaired and checked by reading, not compiled, not run on hardware
//
#include <hip/hip_runtime.h>
#include <cstdint>
#include <cstddef>

// Problem constants (from reference)
#define B_   8
#define N_   1024
#define C_   768
#define H_   12
#define D_   64
#define K3_  2304          // 3*C
#define SCALE_ 0.125f      // 64^-0.5
#define AW_  4             // waves per attention workgroup

typedef __attribute__((ext_vector_type(16))) _Float16 v16h;
typedef __attribute__((ext_vector_type(8)))  _Float16 v8h;
typedef __attribute__((ext_vector_type(8)))  float    v8f;

static __device__ __forceinline__ v16h cat8(v8h a, v8h b) {
  return __builtin_shufflevector(a, b, 0,1,2,3,4,5,6,7,8,9,10,11,12,13,14,15);
}

static __device__ __forceinline__ v8f wmma_f16(v16h a, v16h b, v8f c) {
  return __builtin_amdgcn_wmma_f32_16x16x32_f16(false, a, false, b, (short)0, c,
                                                false, false);
}

// A fragment: 16x32 f16 tile, row-major source, leading dim ldk (halves).
static __device__ __forceinline__ v16h load_a(const _Float16* base, int ldk, int lane) {
  const int m = lane & 15, sel = lane >> 4;
  const _Float16* p = base + (size_t)m * ldk;
  v8h c0 = *reinterpret_cast<const v8h*>(p + sel * 8);
  v8h c1 = *reinterpret_cast<const v8h*>(p + 16 + sel * 8);
  return cat8(c0, c1);
}

// B fragment: 32x16 f16 tile from B^T row-major (row = output column).
static __device__ __forceinline__ v16h load_b(const _Float16* baseT, int ldk, int lane) {
  const int n = lane & 15, ks = lane >> 4;
  const _Float16* p = baseT + (size_t)n * ldk + ks * 8;
  v8h c0 = *reinterpret_cast<const v8h*>(p);
  v8h c1 = *reinterpret_cast<const v8h*>(p + 16);
  return cat8(c0, c1);
}

// ---- gfx1250 async global->LDS copy (ASYNCcnt-tracked) ----
static __device__ __forceinline__ void async_b128(uint32_t lds_byte,
                                                  const void* sbase,
                                                  uint32_t gbyte) {
  asm volatile("global_load_async_to_lds_b128 %0, %1, %2"
               :: "v"(lds_byte), "v"(gbyte), "s"(sbase) : "memory");
}
#define WAIT_ASYNC_0() asm volatile("s_wait_asynccnt 0x0" ::: "memory")
#define WAIT_ASYNC_2() asm volatile("s_wait_asynccnt 0x2" ::: "memory")
#define WAIT_ASYNC_4() asm volatile("s_wait_asynccnt 0x4" ::: "memory")

// Stage one 32-key K chunk (contiguous 4KB) into LDS. 256 x 16B segments,
// 2 per thread across the 128-thread block.
static __device__ __forceinline__ void issue_k_chunk(const _Float16* Kp, int jc,
                                                     const _Float16* KsBuf, int tid) {
#pragma unroll
  for (int k = 0; k < 2; ++k) {
    const int seg = tid + k * 128;
    const uint32_t lds = (uint32_t)(uintptr_t)KsBuf + (uint32_t)seg * 16u;
    const uint32_t goff = (uint32_t)((jc * 2048 + seg * 8) * 2);
    async_b128(lds, Kp, goff);
  }
}

// Stage one 32-key V chunk from VT [64 feats x N]: 64 rows x 64B.
static __device__ __forceinline__ void issue_v_chunk(const _Float16* Vt, int jc,
                                                     const _Float16* VsBuf, int tid) {
#pragma unroll
  for (int k = 0; k < 2; ++k) {
    const int seg = tid + k * 128;          // 0..255
    const int row = seg >> 2, part = seg & 3;
    const uint32_t lds = (uint32_t)(uintptr_t)VsBuf +
                         (uint32_t)((row * 32 + part * 8) * 2);
    const uint32_t goff = (uint32_t)((row * N_ + jc * 32 + part * 8) * 2);
    async_b128(lds, Vt, goff);
  }
}

// Register-blocked 32x64 tile GEMM accumulate: 2 row-frags x 4 col-frags.
static __device__ __forceinline__ void gemm_block_2x4(const _Float16* A,
                                                      const _Float16* Bt, int K,
                                                      int lane, v8f acc[2][4]) {
  for (int kb = 0; kb < K; kb += 32) {
    if (kb + 32 < K) {
      __builtin_prefetch(A + kb + 32, 0, 1);
      __builtin_prefetch(A + (size_t)16 * K + kb + 32, 0, 1);
      __builtin_prefetch(Bt + kb + 32, 0, 1);
      __builtin_prefetch(Bt + (size_t)32 * K + kb + 32, 0, 1);
    }
    v16h a0 = load_a(A + kb, K, lane);
    v16h a1 = load_a(A + (size_t)16 * K + kb, K, lane);
#pragma unroll
    for (int j = 0; j < 4; ++j) {
      v16h b = load_b(Bt + (size_t)j * 16 * K + kb, K, lane);
      acc[0][j] = wmma_f16(a0, b, acc[0][j]);
      acc[1][j] = wmma_f16(a1, b, acc[1][j]);
    }
  }
}

// ---------------- conversion kernels ----------------

__global__ void cvt_f16_kernel(const float* __restrict__ in,
                               _Float16* __restrict__ out, int n) {
  int i = blockIdx.x * blockDim.x + threadIdx.x;
  if (i < n) out[i] = (_Float16)in[i];
}

__global__ void cvt_t_f16_kernel(const float* __restrict__ in,
                                 _Float16* __restrict__ out, int R, int Cc) {
  int i = blockIdx.x * blockDim.x + threadIdx.x;
  if (i < R * Cc) {
    int r = i / Cc, c = i % Cc;
    out[(size_t)c * R + r] = (_Float16)in[i];
  }
}

// ---------------- QKV projection ----------------
__global__ void __launch_bounds__(32)
qkv_gemm_kernel(const _Float16* __restrict__ X, const _Float16* __restrict__ WT,
                _Float16* __restrict__ Qo, _Float16* __restrict__ Ko,
                _Float16* __restrict__ VTo) {
  const int lane = threadIdx.x;
  const int nt = blockIdx.x;   // 0..35
  const int mt = blockIdx.y;   // 0..255

  const v8f vzero = {0.f,0.f,0.f,0.f,0.f,0.f,0.f,0.f};
  v8f acc[2][4] = {{vzero,vzero,vzero,vzero},{vzero,vzero,vzero,vzero}};
  gemm_block_2x4(X + (size_t)mt * 32 * C_, WT + (size_t)nt * 64 * C_, C_, lane, acc);

  const int n = lane & 15, sel = lane >> 4;
  const int col0 = nt * 64;
  const int t = col0 / C_;
  const int h = (col0 % C_) / D_;
  const int row0 = mt * 32;
  const int b    = row0 / N_;
  const int tok0 = row0 % N_;
  const size_t bh = (size_t)b * H_ + h;

#pragma unroll
  for (int j = 0; j < 4; ++j) {
    const int d = j * 16 + n;
#pragma unroll
    for (int i = 0; i < 2; ++i)
#pragma unroll
      for (int e = 0; e < 8; ++e) {
        const int tok = tok0 + i * 16 + sel * 8 + e;
        const float v = acc[i][j][e];
        if (t == 0)      Qo[(bh * N_ + tok) * D_ + d] = (_Float16)(v * SCALE_);
        else if (t == 1) Ko[(bh * N_ + tok) * D_ + d] = (_Float16)v;
        else             VTo[(bh * D_ + d) * N_ + tok] = (_Float16)v;
      }
  }
}

// ---------------- attention ----------------
// 4 waves / WG, 128 queries of one (b,h) per WG. K/V chunks are double-buffered
// in LDS via async global->LDS DMA and shared by all 4 waves.
__global__ void __launch_bounds__(128)
attn_kernel(const _Float16* __restrict__ Qg, const _Float16* __restrict__ Kg,
            const _Float16* __restrict__ VTg, _Float16* __restrict__ Og) {
  __shared__ _Float16 Ks[2][32 * 64];          // 2 x 4KB key chunks (row-major)
  __shared__ _Float16 Vs[2][64 * 32];          // 2 x 4KB V^T chunks [feat][key]
  __shared__ _Float16 Pts[AW_][2][16 * 32];    // per-wave P bounce tiles

  const int tid = threadIdx.x;
  const int lane = tid & 31;
  const int w = tid >> 5;
  const int sel = lane >> 4, r = lane & 15;
  const int qt = blockIdx.x;   // 0..7 (128-query strips)
  const int bhi = blockIdx.y;  // 0..95
  const int b = bhi / H_, h = bhi % H_;

  const _Float16* Q  = Qg  + (size_t)bhi * N_ * D_ + ((size_t)qt * 128 + w * 32) * D_;
  const _Float16* Kp = Kg  + (size_t)bhi * N_ * D_;
  const _Float16* Vt = VTg + (size_t)bhi * D_ * N_;

  // Resident Q fragments: 2 query-tiles x 2 K-chunks (D=64)
  v16h aq[2][2];
#pragma unroll
  for (int qb = 0; qb < 2; ++qb) {
    aq[qb][0] = load_a(Q + (size_t)qb * 16 * D_, D_, lane);
    aq[qb][1] = load_a(Q + (size_t)qb * 16 * D_ + 32, D_, lane);
  }
  const v8f vzero = {0.f,0.f,0.f,0.f,0.f,0.f,0.f,0.f};

  // ---- pass 1: row max over all keys (K chunks streamed through LDS)
  float mx[2][8];
#pragma unroll
  for (int qb = 0; qb < 2; ++qb)
#pragma unroll
    for (int i = 0; i < 8; ++i) mx[qb][i] = -3.0e38f;

  issue_k_chunk(Kp, 0, Ks[0], tid);
  for (int jc = 0; jc < N_ / 32; ++jc) {
    if (jc + 1 < N_ / 32) {
      issue_k_chunk(Kp, jc + 1, Ks[(jc + 1) & 1], tid);
      WAIT_ASYNC_2();
    } else {
      WAIT_ASYNC_0();
    }
    __syncthreads();                              // chunk jc resident
    const _Float16* Kbuf = Ks[jc & 1];
#pragma unroll
    for (int hf = 0; hf < 2; ++hf) {
      const _Float16* Kt = Kbuf + hf * 16 * D_;
      v16h bk0 = load_b(Kt, D_, lane);
      v16h bk1 = load_b(Kt + 32, D_, lane);
#pragma unroll
      for (int qb = 0; qb < 2; ++qb) {
        v8f c = wmma_f16(aq[qb][0], bk0, vzero);
        c = wmma_f16(aq[qb][1], bk1, c);
#pragma unroll
        for (int i = 0; i < 8; ++i) mx[qb][i] = fmaxf(mx[qb][i], c[i]);
      }
    }
    __syncthreads();                              // buffer reusable
  }
#pragma unroll
  for (int qb = 0; qb < 2; ++qb)
#pragma unroll
    for (int i = 0; i < 8; ++i)
      for (int off = 1; off < 16; off <<= 1)
        mx[qb][i] = fmaxf(mx[qb][i], __shfl_xor(mx[qb][i], off, 32));

  // ---- pass 2: recompute S, exponentiate, accumulate O = P@V
  float sum[2][8];
#pragma unroll
  for (int qb = 0; qb < 2; ++qb)
#pragma unroll
    for (int i = 0; i < 8; ++i) sum[qb][i] = 0.f;
  v8f acco[2][4] = {{vzero,vzero,vzero,vzero},{vzero,vzero,vzero,vzero}};

  issue_k_chunk(Kp, 0, Ks[0], tid);
  issue_v_chunk(Vt, 0, Vs[0], tid);
  for (int jc = 0; jc < N_ / 32; ++jc) {
    if (jc + 1 < N_ / 32) {
      issue_k_chunk(Kp, jc + 1, Ks[(jc + 1) & 1], tid);
      issue_v_chunk(Vt, jc + 1, Vs[(jc + 1) & 1], tid);
      WAIT_ASYNC_4();
    } else {
      WAIT_ASYNC_0();
    }
    __syncthreads();                              // chunk jc resident
    const _Float16* Kbuf = Ks[jc & 1];
    const _Float16* Vbuf = Vs[jc & 1];

#pragma unroll
    for (int hf = 0; hf < 2; ++hf) {
      const _Float16* Kt = Kbuf + hf * 16 * D_;
      v16h bk0 = load_b(Kt, D_, lane);
      v16h bk1 = load_b(Kt + 32, D_, lane);
#pragma unroll
      for (int qb = 0; qb < 2; ++qb) {
        v8f c = wmma_f16(aq[qb][0], bk0, vzero);
        c = wmma_f16(aq[qb][1], bk1, c);
#pragma unroll
        for (int i = 0; i < 8; ++i) {
          float e = __expf(c[i] - mx[qb][i]);
          sum[qb][i] += e;
          Pts[w][qb][(sel * 8 + i) * 32 + hf * 16 + r] = (_Float16)e;
        }
      }
    }
    __syncthreads();                              // P tiles visible (DS ordering)

    v16h ap0 = load_a(Pts[w][0], 32, lane);
    v16h ap1 = load_a(Pts[w][1], 32, lane);
#pragma unroll
    for (int ft = 0; ft < 4; ++ft) {              // 4 x 16 output features
      v16h bv = load_b(Vbuf + (size_t)(ft * 16) * 32, 32, lane);
      acco[0][ft] = wmma_f16(ap0, bv, acco[0][ft]);
      acco[1][ft] = wmma_f16(ap1, bv, acco[1][ft]);
    }
    __syncthreads();                              // buffers reusable
  }

  // row sums + normalize + store: out[b, row, h*64 + ft*16 + r]
#pragma unroll
  for (int qb = 0; qb < 2; ++qb) {
#pragma unroll
    for (int i = 0; i < 8; ++i) {
      for (int off = 1; off < 16; off <<= 1)
        sum[qb][i] += __shfl_xor(sum[qb][i], off, 32);
      sum[qb][i] = 1.0f / sum[qb][i];
    }
#pragma unroll
    for (int ft = 0; ft < 4; ++ft)
#pragma unroll
      for (int i = 0; i < 8; ++i) {
        const size_t row = (size_t)b * N_ + qt * 128 + w * 32 + qb * 16 + sel * 8 + i;
        Og[row * C_ + h * D_ + ft * 16 + r] =
            (_Float16)(acco[qb][ft][i] * sum[qb][i]);
      }
  }
}

// ---------------- output projection ----------------
__global__ void __launch_bounds__(32)
out_gemm_kernel(const _Float16* __restrict__ A, const _Float16* __restrict__ WT,
                const float* __restrict__ bias, float* __restrict__ Out) {
  const int lane = threadIdx.x;
  const int nt = blockIdx.x;   // 0..11
  const int mt = blockIdx.y;   // 0..255

  const v8f vzero = {0.f,0.f,0.f,0.f,0.f,0.f,0.f,0.f};
  v8f acc[2][4] = {{vzero,vzero,vzero,vzero},{vzero,vzero,vzero,vzero}};
  gemm_block_2x4(A + (size_t)mt * 32 * C_, WT + (size_t)nt * 64 * C_, C_, lane, acc);

  const int n = lane & 15, sel = lane >> 4;
#pragma unroll
  for (int j = 0; j < 4; ++j) {
    const int col = nt * 64 + j * 16 + n;
    const float bb = bias[col];
#pragma unroll
    for (int i = 0; i < 2; ++i)
#pragma unroll
      for (int e = 0; e < 8; ++e) {
        const int row = mt * 32 + i * 16 + sel * 8 + e;
        Out[(size_t)row * C_ + col] = acc[i][j][e] + bb;
      }
  }
}

// ---------------- launch ----------------
extern "C" void kernel_launch(void* const* d_in, const int* in_sizes, int n_in,
                              void* d_out, int out_size, void* d_ws, size_t ws_size,
                              hipStream_t stream) {
  (void)in_sizes; (void)n_in; (void)out_size; (void)ws_size;
  const float* x      = (const float*)d_in[0];
  const float* w_qkv  = (const float*)d_in[1];
  const float* w_proj = (const float*)d_in[2];
  const float* b_proj = (const float*)d_in[3];
  float* out = (float*)d_out;

  char* ws = (char*)d_ws;
  size_t o = 0;
  _Float16* x16    = (_Float16*)(ws + o); o += (size_t)B_ * N_ * C_ * 2;
  _Float16* wqkvT  = (_Float16*)(ws + o); o += (size_t)K3_ * C_ * 2;
  _Float16* wprojT = (_Float16*)(ws + o); o += (size_t)C_ * C_ * 2;
  _Float16* Qb     = (_Float16*)(ws + o); o += (size_t)B_ * N_ * C_ * 2;
  _Float16* Kb     = (_Float16*)(ws + o); o += (size_t)B_ * N_ * C_ * 2;
  _Float16* VTb    = (_Float16*)(ws + o); o += (size_t)B_ * N_ * C_ * 2;
  _Float16* Ab     = (_Float16*)(ws + o); o += (size_t)B_ * N_ * C_ * 2;

  const int nx = B_ * N_ * C_;
  cvt_f16_kernel<<<(nx + 255) / 256, 256, 0, stream>>>(x, x16, nx);
  cvt_t_f16_kernel<<<(C_ * K3_ + 255) / 256, 256, 0, stream>>>(w_qkv, wqkvT, C_, K3_);
  cvt_t_f16_kernel<<<(C_ * C_ + 255) / 256, 256, 0, stream>>>(w_proj, wprojT, C_, C_);

  qkv_gemm_kernel<<<dim3(K3_ / 64, (B_ * N_) / 32), 32, 0, stream>>>(
      x16, wqkvT, Qb, Kb, VTb);

  attn_kernel<<<dim3(N_ / 128, B_ * H_), 128, 0, stream>>>(Qb, Kb, VTb, Ab);

  out_gemm_kernel<<<dim3(C_ / 64, (B_ * N_) / 32), 32, 0, stream>>>(
      Ab, wprojT, b_proj, out);
}
